// GroupedQueryAttention_79645873537631
// MI455X (gfx1250) — matrix operand order repaired and checked by reading, not compiled, not run on hardware
//
#include <hip/hip_runtime.h>

typedef unsigned short ushort_t;
typedef __attribute__((ext_vector_type(16))) __bf16 v16bf;
typedef __attribute__((ext_vector_type(8)))  float v8f;

union Frag16 { v16bf v; uint4 q[2]; };

// ---------------------------------------------------------------- helpers
__device__ __forceinline__ ushort_t f2bf(float f) {
    union { float f; unsigned u; } x; x.f = f;
    unsigned u = x.u;
    unsigned r = u + 0x7FFFu + ((u >> 16) & 1u);   // round-to-nearest-even
    return (ushort_t)(r >> 16);
}

__device__ __forceinline__ v8f wmma_bf16(v16bf a, v16bf b, v8f c) {
    return __builtin_amdgcn_wmma_f32_16x16x32_bf16(false, a, false, b,
                                                   (short)0, c, false, false);
}

// A-fragment (16x32 bf16): lane holds row lh; chunks [half*8,+8) and [16+half*8,+8)
__device__ __forceinline__ v16bf load_afrag(const ushort_t* rowptr, int half) {
    Frag16 f;
    f.q[0] = *(const uint4*)(rowptr + half * 8);
    f.q[1] = *(const uint4*)(rowptr + half * 8 + 16);
    return f.v;
}

// B-fragment (32x16 bf16): lane holds col lh; chunks [half*8,+8) and [16+half*8,+8)
__device__ __forceinline__ v16bf load_bfrag(const ushort_t* rowptr, int half) {
    Frag16 f;
    f.q[0] = *(const uint4*)(rowptr + half * 8);
    f.q[1] = *(const uint4*)(rowptr + half * 8 + 16);
    return f.v;
}

// async global->LDS copy of 16 bytes per lane (ASYNCcnt-tracked, no VGPR staging)
__device__ __forceinline__ void async_copy_b128(unsigned lds_byte_addr,
                                                const void* gaddr) {
    asm volatile("global_load_async_to_lds_b128 %0, %1, off"
                 :: "v"(lds_byte_addr), "v"(gaddr) : "memory");
}

__device__ __forceinline__ void wait_async0() {
    asm volatile("s_wait_asynccnt 0" ::: "memory");
}

// ---------------------------------------------------------------- cast f32 -> bf16
__global__ void cast_f32_to_bf16(const float* __restrict__ in,
                                 ushort_t* __restrict__ out, int n) {
    for (int i = blockIdx.x * blockDim.x + threadIdx.x; i < n;
         i += gridDim.x * blockDim.x)
        out[i] = f2bf(in[i]);
}

// ---------------------------------------------------------------- V transpose
// Vb: (B, N, 512) row-major with col = hkv*64+dd  ->  Vt: (B, 8, 64, N)
__global__ void vtrans_kernel(const ushort_t* __restrict__ Vb,
                              ushort_t* __restrict__ Vt) {
    const int total = 2 * 8 * 64 * 2048;
    for (int i = blockIdx.x * blockDim.x + threadIdx.x; i < total;
         i += gridDim.x * blockDim.x) {
        int n  = i & 2047;
        int dd = (i >> 11) & 63;
        int hk = (i >> 17) & 7;
        int b  = i >> 20;
        Vt[i] = Vb[((size_t)(b * 2048 + n)) * 512 + hk * 64 + dd];
    }
}

// ---------------------------------------------------------------- GEMM
// C[M x Nout] = A[M x K](bf16, row-major) * W[Nout x K]^T (bf16, row-major) + bias
// block: 256 thr = 8 waves (2x4), tile 128x128, K-step 32,
// LDS double-buffered via GLOBAL_LOAD_ASYNC_TO_LDS_B128 + s_wait_asynccnt
template <bool OUT_F32>
__global__ __launch_bounds__(256)
void gemm_bf16_kernel(const ushort_t* __restrict__ A, const ushort_t* __restrict__ W,
                      const float* __restrict__ bias, void* __restrict__ Cout,
                      int M, int Nout, int K) {
    __shared__ __align__(16) ushort_t sA[2][128][40];
    __shared__ __align__(16) ushort_t sB[2][128][40];

    const int tid = threadIdx.x;
    const int wave = tid >> 5, lane = tid & 31;
    const int lh = lane & 15, half = lane >> 4;
    const int wm = wave >> 2, wn = wave & 3;      // 2 x 4 wave grid

    const int m0 = blockIdx.y * 128;
    const int n0 = blockIdx.x * 128;

    // each thread moves 32 B of A and 32 B of W per K-step (two b128 asyncs each)
    const int lrow = tid >> 1;
    const int lcol = (tid & 1) * 16;
    const ushort_t* gA = A + (size_t)(m0 + lrow) * K + lcol;
    const ushort_t* gB = W + (size_t)(n0 + lrow) * K + lcol;

    const unsigned lA0 = (unsigned)(uintptr_t)&sA[0][lrow][lcol];
    const unsigned lA1 = (unsigned)(uintptr_t)&sA[1][lrow][lcol];
    const unsigned lB0 = (unsigned)(uintptr_t)&sB[0][lrow][lcol];
    const unsigned lB1 = (unsigned)(uintptr_t)&sB[1][lrow][lcol];

    // stage K-tile 0 into buffer 0
    async_copy_b128(lA0,      gA);
    async_copy_b128(lA0 + 16, gA + 8);
    async_copy_b128(lB0,      gB);
    async_copy_b128(lB0 + 16, gB + 8);
    wait_async0();
    __syncthreads();

    v8f acc[4][2];
#pragma unroll
    for (int i = 0; i < 4; ++i)
#pragma unroll
        for (int j = 0; j < 2; ++j) acc[i][j] = {};

    const int nK = K >> 5;
    for (int kt = 0; kt < nK; ++kt) {
        const int cur = kt & 1;
        const bool more = (kt + 1) < nK;
        if (more) {
            // everyone finished reading buffer (1-cur) before the last barrier
            const ushort_t* pa = gA + (size_t)(kt + 1) * 32;
            const ushort_t* pb = gB + (size_t)(kt + 1) * 32;
            const unsigned dA = cur ? lA0 : lA1;
            const unsigned dB = cur ? lB0 : lB1;
            async_copy_b128(dA,      pa);
            async_copy_b128(dA + 16, pa + 8);
            async_copy_b128(dB,      pb);
            async_copy_b128(dB + 16, pb + 8);
        }
        v16bf af[4], bf[2];
#pragma unroll
        for (int i = 0; i < 4; ++i)
            af[i] = load_afrag(&sA[cur][wm * 64 + i * 16 + lh][0], half);
#pragma unroll
        for (int j = 0; j < 2; ++j)
            bf[j] = load_bfrag(&sB[cur][wn * 32 + j * 16 + lh][0], half);
#pragma unroll
        for (int i = 0; i < 4; ++i)
#pragma unroll
            for (int j = 0; j < 2; ++j)
                acc[i][j] = wmma_bf16(af[i], bf[j], acc[i][j]);
        if (more) {
            wait_async0();       // my async copies into next buffer landed
            __syncthreads();     // everyone's did
        }
    }

    // epilogue: C layout row = v + 8*half, col = lh (per 16x16 tile)
#pragma unroll
    for (int j = 0; j < 2; ++j) {
        const int col = n0 + wn * 32 + j * 16 + lh;
        const float bv = bias[col];
#pragma unroll
        for (int i = 0; i < 4; ++i) {
            const int rbase = m0 + wm * 64 + i * 16 + 8 * half;
#pragma unroll
            for (int v = 0; v < 8; ++v) {
                const float val = acc[i][j][v] + bv;
                const size_t idx = (size_t)(rbase + v) * Nout + col;
                if (OUT_F32) ((float*)Cout)[idx] = val;
                else         ((ushort_t*)Cout)[idx] = f2bf(val);
            }
        }
    }
}

// ---------------------------------------------------------------- flash GQA
// grid (32 q-tiles, 64 b*h), block 128 = 4 waves; wave owns 16 Q rows.
__global__ __launch_bounds__(128)
void gqa_flash_kernel(const ushort_t* __restrict__ Qb, const ushort_t* __restrict__ Kb,
                      const ushort_t* __restrict__ Vt, ushort_t* __restrict__ Ob) {
    __shared__ __align__(16) ushort_t sP[4][16][72];

    const int tid = threadIdx.x, wave = tid >> 5, lane = tid & 31;
    const int lh = lane & 15, half = lane >> 4;
    const int bh = blockIdx.y;
    const int b = bh >> 5, h = bh & 31, hkv = h >> 2;
    const int n0 = blockIdx.x * 64 + wave * 16;

    const ushort_t* Qp = Qb + ((size_t)(b * 2048 + n0)) * 2048 + h * 64;
    const ushort_t* Kp = Kb + ((size_t)(b * 2048)) * 512 + hkv * 64;
    const ushort_t* Vp = Vt + ((size_t)(b * 8 + hkv) * 64) * 2048;

    // Q A-fragments for the two 32-wide K chunks of d_head=64
    const ushort_t* qrow = Qp + (size_t)lh * 2048;
    v16bf qf0 = load_afrag(qrow, half);
    v16bf qf1 = load_afrag(qrow + 32, half);

    float m_run[8], l_run[8];
    v8f o[4];
#pragma unroll
    for (int v = 0; v < 8; ++v) { m_run[v] = -1e30f; l_run[v] = 0.0f; }
#pragma unroll
    for (int j = 0; j < 4; ++j) o[j] = {};

    for (int kt = 0; kt < 32; ++kt) {          // 32 tiles of 64 keys
        // warm next tile (global_prefetch_b8)
        if (kt + 1 < 32) {
            __builtin_prefetch(Kp + (size_t)((kt + 1) * 64 + lane * 2) * 512, 0, 1);
            __builtin_prefetch(Vp + (size_t)lane * 2048 + (kt + 1) * 64, 0, 1);
            __builtin_prefetch(Vp + (size_t)(lane + 32) * 2048 + (kt + 1) * 64, 0, 1);
        }
        // ---- scores S (16 x 64), scale = 1/sqrt(64)
        v8f s[4];
#pragma unroll
        for (int c = 0; c < 4; ++c) {
            const ushort_t* kr = Kp + (size_t)(kt * 64 + c * 16 + lh) * 512;
            v8f a = {};
            a = wmma_bf16(qf0, load_bfrag(kr, half), a);
            a = wmma_bf16(qf1, load_bfrag(kr + 32, half), a);
            s[c] = a * 0.125f;
        }
        // ---- online softmax (row = v + 8*half; 16-lane xor reductions)
#pragma unroll
        for (int v = 0; v < 8; ++v) {
            float mx = fmaxf(fmaxf(s[0][v], s[1][v]), fmaxf(s[2][v], s[3][v]));
#pragma unroll
            for (int off = 1; off < 16; off <<= 1)
                mx = fmaxf(mx, __shfl_xor(mx, off, 32));
            const float mnew  = fmaxf(m_run[v], mx);
            const float alpha = __expf(m_run[v] - mnew);
            m_run[v] = mnew;
            float rs = 0.0f;
#pragma unroll
            for (int c = 0; c < 4; ++c) {
                const float p = __expf(s[c][v] - mnew);
                s[c][v] = p;
                rs += p;
            }
#pragma unroll
            for (int off = 1; off < 16; off <<= 1)
                rs += __shfl_xor(rs, off, 32);
            l_run[v] = l_run[v] * alpha + rs;
            o[0][v] *= alpha; o[1][v] *= alpha; o[2][v] *= alpha; o[3][v] *= alpha;
        }
        // ---- C-layout -> A-layout via wave-private LDS
#pragma unroll
        for (int c = 0; c < 4; ++c)
#pragma unroll
            for (int v = 0; v < 8; ++v)
                sP[wave][v + 8 * half][c * 16 + lh] = f2bf(s[c][v]);
        asm volatile("s_wait_dscnt 0" ::: "memory");  // same-wave LDS RAW fence
        const ushort_t* pr = &sP[wave][lh][0];
        v16bf pf0 = load_afrag(pr, half);
        v16bf pf1 = load_afrag(pr + 32, half);
        // ---- O += P @ V   (V^T rows give contiguous-K B fragments)
#pragma unroll
        for (int j = 0; j < 4; ++j) {
            const ushort_t* vr = Vp + (size_t)(j * 16 + lh) * 2048 + kt * 64;
            o[j] = wmma_bf16(pf0, load_bfrag(vr, half), o[j]);
            o[j] = wmma_bf16(pf1, load_bfrag(vr + 32, half), o[j]);
        }
    }

    // ---- normalize + store to (B, N, Hq*64) bf16
#pragma unroll
    for (int v = 0; v < 8; ++v) {
        const float inv = 1.0f / l_run[v];
        const int row = n0 + v + 8 * half;
        const size_t base = ((size_t)(b * 2048 + row)) * 2048 + h * 64;
#pragma unroll
        for (int j = 0; j < 4; ++j)
            Ob[base + j * 16 + lh] = f2bf(o[j][v] * inv);
    }
}

// ---------------------------------------------------------------- launch
extern "C" void kernel_launch(void* const* d_in, const int* in_sizes, int n_in,
                              void* d_out, int out_size, void* d_ws, size_t ws_size,
                              hipStream_t stream) {
    const float* x  = (const float*)d_in[0];
    const float* Wq = (const float*)d_in[1];
    const float* bq = (const float*)d_in[2];
    const float* Wk = (const float*)d_in[3];
    const float* bk = (const float*)d_in[4];
    const float* Wv = (const float*)d_in[5];
    const float* bv = (const float*)d_in[6];
    const float* Wo = (const float*)d_in[7];
    const float* bo = (const float*)d_in[8];

    char* w = (char*)d_ws;
    ushort_t* xb  = (ushort_t*)w; w += (size_t)4096 * 2048 * 2;
    ushort_t* wqb = (ushort_t*)w; w += (size_t)2048 * 2048 * 2;
    ushort_t* wkb = (ushort_t*)w; w += (size_t)512  * 2048 * 2;
    ushort_t* wvb = (ushort_t*)w; w += (size_t)512  * 2048 * 2;
    ushort_t* wob = (ushort_t*)w; w += (size_t)2048 * 2048 * 2;
    ushort_t* qb  = (ushort_t*)w; w += (size_t)4096 * 2048 * 2;
    ushort_t* kb  = (ushort_t*)w; w += (size_t)4096 * 512  * 2;
    ushort_t* vb  = (ushort_t*)w; w += (size_t)4096 * 512  * 2;
    ushort_t* vt  = (ushort_t*)w; w += (size_t)4096 * 512  * 2;
    ushort_t* ao  = (ushort_t*)w; w += (size_t)4096 * 2048 * 2;

    cast_f32_to_bf16<<<2048, 256, 0, stream>>>(x,  xb,  4096 * 2048);
    cast_f32_to_bf16<<<1024, 256, 0, stream>>>(Wq, wqb, 2048 * 2048);
    cast_f32_to_bf16<<<512,  256, 0, stream>>>(Wk, wkb, 512 * 2048);
    cast_f32_to_bf16<<<512,  256, 0, stream>>>(Wv, wvb, 512 * 2048);
    cast_f32_to_bf16<<<1024, 256, 0, stream>>>(Wo, wob, 2048 * 2048);

    // Q, K, V projections
    gemm_bf16_kernel<false><<<dim3(16, 32), 256, 0, stream>>>(xb, wqb, bq, qb, 4096, 2048, 2048);
    gemm_bf16_kernel<false><<<dim3(4,  32), 256, 0, stream>>>(xb, wkb, bk, kb, 4096, 512, 2048);
    gemm_bf16_kernel<false><<<dim3(4,  32), 256, 0, stream>>>(xb, wvb, bv, vb, 4096, 512, 2048);

    vtrans_kernel<<<1024, 256, 0, stream>>>(vb, vt);

    gqa_flash_kernel<<<dim3(32, 64), 128, 0, stream>>>(qb, kb, vt, ao);

    // output projection (fp32 out)
    gemm_bf16_kernel<true><<<dim3(16, 32), 256, 0, stream>>>(ao, wob, bo, (float*)d_out,
                                                             4096, 2048, 2048);
}
